// LSTM_78176994722453
// MI455X (gfx1250) — matrix operand order repaired and checked by reading, not compiled, not run on hardware
//
#include <hip/hip_runtime.h>
#include <hip/hip_bf16.h>
#include <math.h>

typedef __attribute__((ext_vector_type(16))) __bf16 v16bf;
typedef __attribute__((ext_vector_type(8)))  __bf16 v8bf;
typedef __attribute__((ext_vector_type(8)))  float  v8f;
typedef __attribute__((ext_vector_type(4)))  int    v4i;

#define Bsz 64
#define Tsz 512
#define Dsz 1024
#define Nsz 1024
#define NBLK 32
#define LROW 72              // LDS row stride in bf16 (64 data + 8 pad), 144B
#define LBUF (64 * LROW)     // one LDS A-buffer

#if __has_builtin(__builtin_amdgcn_global_load_async_to_lds_b128)
#define USE_ASYNC_LDS 1
#else
#define USE_ASYNC_LDS 0
#endif

typedef __attribute__((address_space(1))) v4i gas_v4i;
typedef __attribute__((address_space(3))) v4i las_v4i;

__device__ __forceinline__ float sigf(float x) { return 1.0f / (1.0f + __expf(-x)); }

__device__ __forceinline__ void async_wait_all()
{
#if USE_ASYNC_LDS
#if __has_builtin(__builtin_amdgcn_s_wait_asynccnt)
    __builtin_amdgcn_s_wait_asynccnt(0);
#else
    asm volatile("s_wait_asynccnt 0x0" ::: "memory");
#endif
#endif
}

// stage 16 bytes (8 bf16) from global to LDS
__device__ __forceinline__ void stage16(const __bf16* src, __bf16* dst)
{
#if USE_ASYNC_LDS
    __builtin_amdgcn_global_load_async_to_lds_b128(
        (gas_v4i*)const_cast<__bf16*>(src), (las_v4i*)dst, 0, 0);
#else
    *(v8bf*)dst = *(const v8bf*)src;
#endif
}

// device-wide sync: monotonic arrival counter, no reset race
__device__ __forceinline__ void gsync(unsigned* cnt, unsigned target)
{
    __threadfence();
    __syncthreads();
    if (threadIdx.x == 0) {
        __hip_atomic_fetch_add(cnt, 1u, __ATOMIC_ACQ_REL, __HIP_MEMORY_SCOPE_AGENT);
        while (__hip_atomic_load(cnt, __ATOMIC_ACQUIRE, __HIP_MEMORY_SCOPE_AGENT) < target)
            __builtin_amdgcn_s_sleep(1);
    }
    __syncthreads();
    __threadfence();
}

// ---------------------------------------------------------------------------
// One-time: convert x [B,T,D] fp32 -> bf16
// ---------------------------------------------------------------------------
__global__ __launch_bounds__(256) void cvt_x(const float* __restrict__ x,
                                             __bf16* __restrict__ xbf)
{
    size_t i = ((size_t)blockIdx.x * 256 + threadIdx.x) * 8;
    const float4 f0 = *(const float4*)(x + i);
    const float4 f1 = *(const float4*)(x + i + 4);
    v8bf v;
    v[0] = (__bf16)f0.x; v[1] = (__bf16)f0.y; v[2] = (__bf16)f0.z; v[3] = (__bf16)f0.w;
    v[4] = (__bf16)f1.x; v[5] = (__bf16)f1.y; v[6] = (__bf16)f1.z; v[7] = (__bf16)f1.w;
    *(v8bf*)(xbf + i) = v;
}

// ---------------------------------------------------------------------------
// Pack weights: WRT = transposed bf16 [4096 cols][2048 k]  (k<1024: W rows,
// k>=1024: R rows; gate order z,r,o,h with R_h block zeroed).
// RHT = transposed bf16 [1024][1024] of R_h.
// ---------------------------------------------------------------------------
__global__ __launch_bounds__(256) void pack_weights(
    const float* __restrict__ Wz, const float* __restrict__ Wr,
    const float* __restrict__ Wo, const float* __restrict__ Wh,
    const float* __restrict__ Rz, const float* __restrict__ Rr,
    const float* __restrict__ Ro, const float* __restrict__ Rh,
    __bf16* __restrict__ WRT, __bf16* __restrict__ RHT)
{
    size_t i = (size_t)blockIdx.x * 256 + threadIdx.x;
    const size_t wrtN = (size_t)4096 * 2048;
    if (i < wrtN) {
        int n = (int)(i >> 11);
        int k = (int)(i & 2047);
        int g = n >> 10, nn = n & 1023;
        float v;
        if (k < 1024) {
            const float* W = (g == 0) ? Wz : (g == 1) ? Wr : (g == 2) ? Wo : Wh;
            v = W[(size_t)k * Nsz + nn];
        } else {
            int kk = k - 1024;
            v = (g == 0) ? Rz[(size_t)kk * Nsz + nn]
              : (g == 1) ? Rr[(size_t)kk * Nsz + nn]
              : (g == 2) ? Ro[(size_t)kk * Nsz + nn]
              : 0.0f;
        }
        WRT[i] = (__bf16)v;
    } else {
        size_t j = i - wrtN;
        if (j < (size_t)1024 * 1024) {
            int n = (int)(j >> 10), k = (int)(j & 1023);
            RHT[j] = (__bf16)Rh[(size_t)k * Nsz + n];
        }
    }
}

__global__ __launch_bounds__(256) void init_state(__bf16* __restrict__ h,
                                                  float* __restrict__ c,
                                                  unsigned* __restrict__ syncc)
{
    int i = blockIdx.x * 256 + threadIdx.x;   // 65536 = 64*1024
    h[i] = (__bf16)0.0f;
    c[i] = 0.0f;
    if (i == 0) *syncc = 0u;
}

// ---------------------------------------------------------------------------
// GEMM body with fused epilogues; 4 waves x (two 16-col strips x 4 M-tiles).
// 64-wide K chunks, LDS double buffer, async B128 staging, 1 barrier / chunk.
// MODE 0 (gates): acc=0 -> pre[64,4096]; r-cols also emit rh=sig(acc+Br)*h.
// MODE 1 (hbar):  acc preloaded from pre[:,3072+col]; += rh@R_h; full update.
// ---------------------------------------------------------------------------
template <int MODE>
__device__ __forceinline__ void gemm_body(
    __bf16* At,                                       // LDS, 2 buffers of LBUF
    const __bf16* __restrict__ a0, int ld0, int kx,
    const __bf16* __restrict__ a1, int kh,
    const __bf16* __restrict__ BT, int ldbt, int colblk,
    float* __restrict__ pre,
    const __bf16* __restrict__ hin, __bf16* __restrict__ hout,
    __bf16* __restrict__ rh,
    const float* __restrict__ Bz, const float* __restrict__ Br,
    const float* __restrict__ Bo, const float* __restrict__ Bh,
    float* __restrict__ c, float* __restrict__ out, int last)
{
    const int tid  = threadIdx.x;
    const int lane = tid & 31;
    const int wv   = tid >> 5;        // 0..3
    const int hs   = lane >> 4;
    const int ml   = lane & 15;
    const int nbase = colblk + wv * 32;
    const __bf16* Bcol0 = BT + (size_t)(nbase + ml) * ldbt;
    const __bf16* Bcol1 = BT + (size_t)(nbase + 16 + ml) * ldbt;
    const int ktot = kx + kh;

    const int sr  = tid >> 1;         // staging row 0..63
    const int sc0 = (tid & 1) << 5;   // staging col group 0 or 32 (4 x B128)

    v8f acc[2][4];
    if (MODE == 1) {
#pragma unroll
        for (int s = 0; s < 2; ++s) {
            const int col = nbase + s * 16 + ml;
#pragma unroll
            for (int mt = 0; mt < 4; ++mt) {
                const float* op = pre + (size_t)(mt * 16 + 8 * hs) * 4096 + 3072 + col;
#pragma unroll
                for (int i = 0; i < 8; ++i) acc[s][mt][i] = op[(size_t)i * 4096];
            }
        }
    } else {
#pragma unroll
        for (int s = 0; s < 2; ++s)
#pragma unroll
            for (int mt = 0; mt < 4; ++mt)
                acc[s][mt] = (v8f){0.f, 0.f, 0.f, 0.f, 0.f, 0.f, 0.f, 0.f};
    }

    {   // prologue: stage chunk 0 into buffer 0
        const int kk = sc0;
        const __bf16* src = (kk < kx) ? (a0 + (size_t)sr * ld0 + kk)
                                      : (a1 + (size_t)sr * kh + (kk - kx));
        __bf16* dst = At + sr * LROW + sc0;
        stage16(src, dst);           stage16(src + 8, dst + 8);
        stage16(src + 16, dst + 16); stage16(src + 24, dst + 24);
    }

    int cur = 0;
    for (int k0 = 0; k0 < ktot; k0 += 64) {
        async_wait_all();
        __syncthreads();

        if (k0 + 64 < ktot) {
            const int kk = k0 + 64 + sc0;
            const __bf16* src = (kk < kx) ? (a0 + (size_t)sr * ld0 + kk)
                                          : (a1 + (size_t)sr * kh + (kk - kx));
            __bf16* dst = At + (cur ^ 1) * LBUF + sr * LROW + sc0;
            stage16(src, dst);           stage16(src + 8, dst + 8);
            stage16(src + 16, dst + 16); stage16(src + 24, dst + 24);
            __builtin_prefetch(Bcol0 + k0 + 64 + 16 * hs, 0, 0);
            __builtin_prefetch(Bcol1 + k0 + 64 + 16 * hs, 0, 0);
        }

#pragma unroll
        for (int cc = 0; cc < 2; ++cc) {          // two 32-K sub-chunks
            const int kb = k0 + cc * 32;
            const v16bf bf0 = *(const v16bf*)(Bcol0 + kb + 16 * hs);
            const v16bf bf1 = *(const v16bf*)(Bcol1 + kb + 16 * hs);
#pragma unroll
            for (int mt = 0; mt < 4; ++mt) {
                const __bf16* ap = At + cur * LBUF + (mt * 16 + ml) * LROW
                                 + cc * 32 + 16 * hs;
                const v8bf p0 = *(const v8bf*)ap;
                const v8bf p1 = *(const v8bf*)(ap + 8);
                const v16bf afrag = __builtin_shufflevector(
                    p0, p1, 0, 1, 2, 3, 4, 5, 6, 7, 8, 9, 10, 11, 12, 13, 14, 15);
                acc[0][mt] = __builtin_amdgcn_wmma_f32_16x16x32_bf16(
                    false, afrag, false, bf0, (short)0, acc[0][mt], false, false);
                acc[1][mt] = __builtin_amdgcn_wmma_f32_16x16x32_bf16(
                    false, afrag, false, bf1, (short)0, acc[1][mt], false, false);
            }
        }
        cur ^= 1;
    }

    if (MODE == 0) {
#pragma unroll
        for (int s = 0; s < 2; ++s) {
            const int col = nbase + s * 16 + ml;
            const bool isR = (col >= 1024) && (col < 2048);
#pragma unroll
            for (int mt = 0; mt < 4; ++mt) {
#pragma unroll
                for (int i = 0; i < 8; ++i) {
                    const int row = mt * 16 + i + 8 * hs;
                    const float v = acc[s][mt][i];
                    pre[(size_t)row * 4096 + col] = v;
                    if (isR) {
                        const int n = col - 1024;
                        const size_t si = (size_t)row * 1024 + n;
                        const float r = sigf(v + Br[n]);
                        rh[si] = (__bf16)(r * (float)hin[si]);
                    }
                }
            }
        }
    } else {
#pragma unroll
        for (int s = 0; s < 2; ++s) {
            const int col = nbase + s * 16 + ml;   // 0..1023
#pragma unroll
            for (int mt = 0; mt < 4; ++mt) {
#pragma unroll
                for (int i = 0; i < 8; ++i) {
                    const int row = mt * 16 + i + 8 * hs;
                    const float* pr = pre + (size_t)row * 4096;
                    const size_t si = (size_t)row * 1024 + col;
                    const float hb = tanhf(acc[s][mt][i] + Bh[col]);
                    const float z  = sigf(pr[col]        + Bz[col]);
                    const float r  = sigf(pr[1024 + col] + Br[col]);
                    const float o  = sigf(pr[2048 + col] + Bo[col]);
                    const float cn = z * c[si] + r * hb;
                    c[si] = cn;
                    const float y = o * tanhf(cn);
                    hout[si] = (__bf16)y;
                    if (last) out[si] = y;
                }
            }
        }
    }
}

// ---------------------------------------------------------------------------
// Persistent kernel: runs all T=512 steps; device-wide sync between phases.
// Phase A: all 32 blocks (4096 gate cols).  Phase C: blocks 0..7 (1024 cols).
// ---------------------------------------------------------------------------
__global__ __launch_bounds__(128) void lstm_persistent(
    const __bf16* __restrict__ xbf, const __bf16* __restrict__ WRT,
    const __bf16* __restrict__ RHT,
    float* __restrict__ pre, __bf16* __restrict__ h, __bf16* __restrict__ rh,
    float* __restrict__ c,
    const float* __restrict__ Bz, const float* __restrict__ Br,
    const float* __restrict__ Bo, const float* __restrict__ Bh,
    float* __restrict__ out, unsigned* __restrict__ syncc)
{
    __shared__ __attribute__((aligned(16))) __bf16 At[2 * LBUF];
    unsigned target = 0;
    for (int t = 0; t < Tsz; ++t) {
        gemm_body<0>(At, xbf + (size_t)t * Dsz, Tsz * Dsz, 1024, h, 1024,
                     WRT, 2048, blockIdx.x * 128,
                     pre, h, h, rh, Bz, Br, Bo, Bh, c, out, 0);
        target += NBLK;
        gsync(syncc, target);
        if (blockIdx.x < 8)
            gemm_body<1>(At, (const __bf16*)nullptr, 0, 0, rh, 1024,
                         RHT, 1024, blockIdx.x * 128,
                         pre, h, h, rh, Bz, Br, Bo, Bh, c, out, t == Tsz - 1);
        target += NBLK;
        gsync(syncc, target);
    }
}

extern "C" void kernel_launch(void* const* d_in, const int* in_sizes, int n_in,
                              void* d_out, int out_size, void* d_ws, size_t ws_size,
                              hipStream_t stream)
{
    (void)in_sizes; (void)n_in; (void)out_size; (void)ws_size;
    const float* x  = (const float*)d_in[0];
    const float* Wz = (const float*)d_in[1];
    const float* Wr = (const float*)d_in[2];
    const float* Wh = (const float*)d_in[3];
    const float* Wo = (const float*)d_in[4];
    const float* Rz = (const float*)d_in[5];
    const float* Rr = (const float*)d_in[6];
    const float* Rh = (const float*)d_in[7];
    const float* Ro = (const float*)d_in[8];
    const float* Bz = (const float*)d_in[9];
    const float* Br = (const float*)d_in[10];
    const float* Bh = (const float*)d_in[11];
    const float* Bo = (const float*)d_in[12];

    char* ws = (char*)d_ws;
    size_t off = 0;
    __bf16* WRT = (__bf16*)(ws + off); off += (size_t)4096 * 2048 * 2;  // 16 MB
    __bf16* RHT = (__bf16*)(ws + off); off += (size_t)1024 * 1024 * 2;  //  2 MB
    __bf16* h   = (__bf16*)(ws + off); off += (size_t)64 * 1024 * 2;
    __bf16* rh  = (__bf16*)(ws + off); off += (size_t)64 * 1024 * 2;
    float*  c   = (float*) (ws + off); off += (size_t)64 * 1024 * 4;
    float*  pre = (float*) (ws + off); off += (size_t)64 * 4096 * 4;
    unsigned* syncc = (unsigned*)(ws + off); off += 256;
    __bf16* xbf = (__bf16*)(ws + off); off += (size_t)Bsz * Tsz * Dsz * 2; // 64 MB

    {
        size_t tot = (size_t)4096 * 2048 + (size_t)1024 * 1024;
        pack_weights<<<(int)((tot + 255) / 256), 256, 0, stream>>>(
            Wz, Wr, Wo, Wh, Rz, Rr, Ro, Rh, WRT, RHT);
    }
    cvt_x<<<(int)(((size_t)Bsz * Tsz * Dsz / 8 + 255) / 256), 256, 0, stream>>>(x, xbf);
    init_state<<<256, 256, 0, stream>>>(h, c, syncc);

    lstm_persistent<<<NBLK, 128, 0, stream>>>(
        xbf, WRT, RHT, pre, h, rh, c, Bz, Br, Bo, Bh, (float*)d_out, syncc);
}
